// CausalMLA_42554535969107
// MI455X (gfx1250) — matrix operand order repaired and checked by reading, not compiled, not run on hardware
//
#include <hip/hip_runtime.h>
#include <hip/hip_bf16.h>
#include <string.h>

#define B_   2
#define L_   2048
#define D_   2048
#define H_   16
#define HD_  128
#define LAT_ 32

typedef __attribute__((ext_vector_type(16))) __bf16 bf16x16;
typedef __attribute__((ext_vector_type(8)))  float  f32x8;
typedef __attribute__((ext_vector_type(4)))  unsigned tdm_g0_t;
typedef __attribute__((ext_vector_type(8)))  int      tdm_g1_t;
typedef __attribute__((ext_vector_type(4)))  int      tdm_g2_t;
typedef __attribute__((ext_vector_type(8)))  int      tdm_g4_t;

// ---------- bf16 <-> f32 helpers ---------------------------------------------
__device__ __forceinline__ float bf2f(__bf16 x) {
    unsigned short u; __builtin_memcpy(&u, &x, 2);
    unsigned v = (unsigned)u << 16;
    float f; __builtin_memcpy(&f, &v, 4);
    return f;
}
__device__ __forceinline__ __bf16 f2bf(float f) {
    unsigned v; __builtin_memcpy(&v, &f, 4);
    unsigned r = (v + 0x7FFFu + ((v >> 16) & 1u)) >> 16;   // RNE
    unsigned short u = (unsigned short)r;
    __bf16 b; __builtin_memcpy(&b, &u, 2);
    return b;
}

// ---------- CDNA5 async global->LDS copy (ASYNCcnt path) ----------------------
// Generic LDS addresses carry the LDS byte offset in their low 32 bits
// (flat aperture spec).
__device__ __forceinline__ void gld_async_b128(const void* g, void* lds) {
    unsigned l = (unsigned)(uintptr_t)lds;
    asm volatile("global_load_async_to_lds_b128 %0, %1, off"
                 :: "v"(l), "v"(g) : "memory");
}
__device__ __forceinline__ void wait_async0() {
    asm volatile("s_wait_asynccnt 0x0" ::: "memory");
}

// ---------- WMMA fragment loaders --------------------------------------------
// A-matrix 16x32 bf16 (split-K layout): lane<16 holds row=lane, K 0..7 & 16..23;
// lane>=16 holds row=lane-16, K 8..15 & 24..31.
__device__ __forceinline__ bf16x16 load_frag_a(const __bf16* base, int ld) {
    int lane = threadIdx.x & 31;
    int row  = lane & 15;
    int c0   = (lane >> 4) << 3;
    const __bf16* p = base + row * ld + c0;
    union { bf16x16 v; uint4 u[2]; } f;
    f.u[0] = *(const uint4*)(p);
    f.u[1] = *(const uint4*)(p + 16);
    return f.v;
}
// B-matrix 32x16 bf16 (N x K source rows, split-K layout): lane<16 holds
// col n=lane, K 0..7 & 16..23; lane>=16 holds col n=lane-16, K 8..15 & 24..31.
__device__ __forceinline__ bf16x16 load_frag_b(const __bf16* base, int ld) {
    int lane = threadIdx.x & 31;
    int row  = lane & 15;
    int c0   = (lane >> 4) << 3;
    const __bf16* p = base + row * ld + c0;
    union { bf16x16 v; uint4 u[2]; } f;
    f.u[0] = *(const uint4*)(p);
    f.u[1] = *(const uint4*)(p + 16);
    return f.v;
}
__device__ __forceinline__ f32x8 wmma_bf16(bf16x16 a, bf16x16 b, f32x8 c) {
    return __builtin_amdgcn_wmma_f32_16x16x32_bf16(false, a, false, b,
                                                   (short)0, c, false, false);
}

// ---------- K0: fp32 -> bf16 convert -----------------------------------------
__global__ void k_convert(const float* __restrict__ src, __bf16* __restrict__ dst, int n) {
    for (int i = blockIdx.x * blockDim.x + threadIdx.x; i < n; i += gridDim.x * blockDim.x)
        dst[i] = f2bf(src[i]);
}

// ---------- K1/K5: Y[M,N] = X[M,K] * W[N,K]^T + bias --------------------------
// Double-buffered: async global->LDS fill of tile k+1 overlaps WMMA on tile k.
template<bool OUT_F32>
__global__ void __launch_bounds__(256)
k_gemm(const __bf16* __restrict__ X, const __bf16* __restrict__ W,
       const float* __restrict__ bias, void* __restrict__ Y,
       int M, int N, int K) {
    __shared__ __bf16 As[2][128 * 32];
    __shared__ __bf16 Bs[2][128 * 32];
    const int bm = blockIdx.x * 128;
    const int bn = blockIdx.y * 128;
    const int tid = threadIdx.x, lane = tid & 31, wave = tid >> 5;
    const int wm = (wave & 3) * 32;     // 4 waves along M (32 rows each)
    const int wn = (wave >> 2) * 64;    // 2 waves along N (64 cols each)

    auto stage = [&](int buf, int kk) {
        #pragma unroll
        for (int p = 0; p < 2; ++p) {
            int lin = tid + p * 256;                 // 512 x 16B chunks per matrix
            int row = lin >> 2;
            int col = (lin & 3) * 8;
            gld_async_b128(&X[(size_t)(bm + row) * K + kk + col], &As[buf][row * 32 + col]);
            gld_async_b128(&W[(size_t)(bn + row) * K + kk + col], &Bs[buf][row * 32 + col]);
        }
    };

    f32x8 acc[2][4] = {};
    stage(0, 0);
    wait_async0();
    __syncthreads();
    for (int kk = 0; kk < K; kk += 32) {
        const int cur = (kk >> 5) & 1;
        if (kk + 32 < K) stage(cur ^ 1, kk + 32);    // overlap next fill with math
        bf16x16 af[2], bfr[4];
        #pragma unroll
        for (int i = 0; i < 2; ++i) af[i]  = load_frag_a(&As[cur][(wm + i * 16) * 32], 32);
        #pragma unroll
        for (int j = 0; j < 4; ++j) bfr[j] = load_frag_b(&Bs[cur][(wn + j * 16) * 32], 32);
        #pragma unroll
        for (int i = 0; i < 2; ++i)
            #pragma unroll
            for (int j = 0; j < 4; ++j)
                acc[i][j] = wmma_bf16(af[i], bfr[j], acc[i][j]);
        wait_async0();
        __syncthreads();
    }
    int n0 = lane & 15, mb = (lane >> 4) * 8;
    #pragma unroll
    for (int i = 0; i < 2; ++i)
        #pragma unroll
        for (int j = 0; j < 4; ++j) {
            int gn = bn + wn + j * 16 + n0;
            float bv = bias[gn];
            #pragma unroll
            for (int r = 0; r < 8; ++r) {
                int gm = bm + wm + i * 16 + mb + r;
                float val = acc[i][j][r] + bv;
                if (OUT_F32) ((float*)Y)[(size_t)gm * N + gn] = val;
                else         ((__bf16*)Y)[(size_t)gm * N + gn] = f2bf(val);
            }
        }
}

// ---------- K2: (RoPE +) latent compression  out = rope(q_head) @ Wlat^T + b --
// Weight tile (32x128 bf16 = 8KB) is fetched by the Tensor Data Mover: one
// 1-D D# (count=1, data_size=2B, tile_dim0=4096) issued by wave 0.
__global__ void __launch_bounds__(256)
k_compress(const __bf16* __restrict__ QKV,     // [B*L, D]
           const float*  __restrict__ freqs,   // [L, HD/2, 2]
           const __bf16* __restrict__ Wlat,    // [LAT, HD]
           const float*  __restrict__ blat,    // [LAT]
           __bf16* __restrict__ out,           // [B*H*L, LAT]
           int doRope) {
    __shared__ __bf16 Qs[128 * HD_];
    __shared__ __bf16 Ws[LAT_ * HD_];
    const int bh = blockIdx.y, b = bh / H_, h = bh % H_;
    const int l0 = blockIdx.x * 128;
    const int tid = threadIdx.x, lane = tid & 31, wave = tid >> 5;

    if (wave == 0) {                                 // TDM weight fetch
        const unsigned NEL = LAT_ * HD_;             // 4096 elements
        unsigned lds = (unsigned)(uintptr_t)(void*)&Ws[0];
        unsigned long long ga = (unsigned long long)(uintptr_t)Wlat;
        tdm_g0_t g0;
        g0[0] = 1u;                                  // count=1, user mode
        g0[1] = lds;                                 // lds_addr
        g0[2] = (unsigned)(ga & 0xFFFFFFFFu);        // global_addr[31:0]
        g0[3] = (unsigned)((ga >> 32) & 0x01FFFFFFu) // global_addr[56:32]
                | (2u << 30);                        // type=2 ("image")
        tdm_g1_t g1 = {};
        g1[0] = (int)(1u << 16);                     // wgmask=0, data_size=1 (2B)
        g1[1] = (int)((NEL & 0xFFFFu) << 16);        // tensor_dim0[15:0]
        g1[2] = (int)((NEL >> 16) | (1u << 16));     // tensor_dim0[31:16], tensor_dim1=1
        g1[3] = (int)((NEL & 0xFFFFu) << 16);        // tile_dim0=4096
        g1[4] = 0;                                   // tile_dim1=0 (1-D), tile_dim2=0
        g1[5] = (int)NEL;                            // tensor_dim0_stride[31:0]
        g1[6] = 0; g1[7] = 0;
        tdm_g2_t gz = {};
        tdm_g4_t gz8 = {};
        __builtin_amdgcn_tensor_load_to_lds(g0, g1, gz, gz, gz8, 0);
        __builtin_amdgcn_s_wait_tensorcnt(0);
    }
    #pragma unroll
    for (int p = 0; p < 8; ++p) {                    // 128x128 activation tile + RoPE
        int lin = tid + p * 256;
        int row = lin >> 4;
        int col = (lin & 15) * 8;
        uint4 raw = *(const uint4*)&QKV[(size_t)(b * L_ + l0 + row) * D_ + h * HD_ + col];
        __bf16 e[8]; __builtin_memcpy(e, &raw, 16);
        if (doRope) {
            int l = l0 + row;
            #pragma unroll
            for (int j = 0; j < 4; ++j) {
                int i2 = (col >> 1) + j;
                float cr = freqs[(l * (HD_ / 2) + i2) * 2 + 0];
                float ci = freqs[(l * (HD_ / 2) + i2) * 2 + 1];
                float a = bf2f(e[2 * j]), bb = bf2f(e[2 * j + 1]);
                e[2 * j]     = f2bf(a * cr - bb * ci);
                e[2 * j + 1] = f2bf(a * ci + bb * cr);
            }
        }
        __builtin_memcpy(&Qs[row * HD_ + col], e, 16);
    }
    __syncthreads();
    f32x8 acc[2] = {};
    for (int ks = 0; ks < HD_; ks += 32) {
        bf16x16 a = load_frag_a(&Qs[(wave * 16) * HD_ + ks], HD_);
        #pragma unroll
        for (int j = 0; j < 2; ++j) {
            bf16x16 w = load_frag_b(&Ws[(j * 16) * HD_ + ks], HD_);
            acc[j] = wmma_bf16(a, w, acc[j]);
        }
    }
    int n0 = lane & 15, mb = (lane >> 4) * 8;
    #pragma unroll
    for (int j = 0; j < 2; ++j) {
        int c = j * 16 + n0;
        float bv = blat[c];
        #pragma unroll
        for (int r = 0; r < 8; ++r) {
            int l = l0 + wave * 16 + mb + r;
            out[((size_t)bh * L_ + l) * LAT_ + c] = f2bf(acc[j][r] + bv);
        }
    }
}

// ---------- K3: causal flash attention over latent dim 32 ---------------------
__global__ void __launch_bounds__(256)
k_attn(const __bf16* __restrict__ qc, const __bf16* __restrict__ kc,
       const __bf16* __restrict__ vc, __bf16* __restrict__ out) {
    __shared__ __bf16 Ks[128 * LAT_];      // [key][lat]
    __shared__ __bf16 VsT[LAT_ * 128];     // [lat][key]  (transposed stage)
    __shared__ __bf16 Ps[8][16 * LAT_];    // per-wave P scratch
    const int bh = blockIdx.y;
    const int q0 = blockIdx.x * 128;
    const __bf16* Q  = qc + (size_t)bh * L_ * LAT_;
    const __bf16* Kp = kc + (size_t)bh * L_ * LAT_;
    const __bf16* Vp = vc + (size_t)bh * L_ * LAT_;
    const int tid = threadIdx.x, lane = tid & 31, wave = tid >> 5;
    const int qrowBase = q0 + wave * 16;
    const int n0 = lane & 15, mb = (lane >> 4) * 8;
    const float scale = 0.17677669529663687f;       // 1/sqrt(32)

    bf16x16 qf = load_frag_a(&Q[(size_t)qrowBase * LAT_], LAT_);
    f32x8 o0 = {}, o1 = {};
    float mr[8], lr[8];
    #pragma unroll
    for (int r = 0; r < 8; ++r) { mr[r] = -1e30f; lr[r] = 0.f; }

    const int nChunks = blockIdx.x + 1;
    for (int ch = 0; ch < nChunks; ++ch) {
        const int k0 = ch * 128;
        __syncthreads();
        #pragma unroll
        for (int p = 0; p < 2; ++p) {
            int lin = tid + p * 256;
            int row = lin >> 2, col = (lin & 3) * 8;
            // K chunk: fixed-function async copy, no VGPR round-trip
            gld_async_b128(&Kp[(size_t)(k0 + row) * LAT_ + col], &Ks[row * LAT_ + col]);
            // V chunk: needs transpose -> stays on the VGPR path
            uint4 vraw = *(const uint4*)&Vp[(size_t)(k0 + row) * LAT_ + col];
            __bf16 e[8]; __builtin_memcpy(e, &vraw, 16);
            #pragma unroll
            for (int j = 0; j < 8; ++j) VsT[(col + j) * 128 + row] = e[j];
        }
        wait_async0();
        __syncthreads();
        for (int s = 0; s < 128; s += 32) {
            f32x8 s0 = {}, s1 = {};
            s0 = wmma_bf16(qf, load_frag_b(&Ks[(s +  0) * LAT_], LAT_), s0);
            s1 = wmma_bf16(qf, load_frag_b(&Ks[(s + 16) * LAT_], LAT_), s1);
            const int key0 = k0 + s + n0, key1 = key0 + 16;
            #pragma unroll
            for (int r = 0; r < 8; ++r) {
                const int qrow = qrowBase + mb + r;
                float a = (key0 <= qrow) ? s0[r] * scale : -1e30f;
                float b = (key1 <= qrow) ? s1[r] * scale : -1e30f;
                float t = fmaxf(a, b);
                #pragma unroll
                for (int off = 1; off < 16; off <<= 1)
                    t = fmaxf(t, __shfl_xor(t, off, 32));      // stays in 16-lane half
                float mn = fmaxf(mr[r], t);
                float alpha = __expf(mr[r] - mn);
                mr[r] = mn;
                float p0 = __expf(a - mn);                     // masked -> 0 exactly
                float p1 = __expf(b - mn);
                float rs = p0 + p1;
                #pragma unroll
                for (int off = 1; off < 16; off <<= 1)
                    rs += __shfl_xor(rs, off, 32);
                lr[r] = lr[r] * alpha + rs;
                o0[r] *= alpha; o1[r] *= alpha;
                Ps[wave][(mb + r) * LAT_ + n0]      = f2bf(p0);
                Ps[wave][(mb + r) * LAT_ + 16 + n0] = f2bf(p1);
            }
            bf16x16 pf  = load_frag_a(&Ps[wave][0], LAT_);
            bf16x16 vf0 = load_frag_b(&VsT[ 0 * 128 + s], 128);
            bf16x16 vf1 = load_frag_b(&VsT[16 * 128 + s], 128);
            o0 = wmma_bf16(pf, vf0, o0);
            o1 = wmma_bf16(pf, vf1, o1);
        }
    }
    #pragma unroll
    for (int r = 0; r < 8; ++r) {
        float inv = 1.f / lr[r];
        size_t rowoff = ((size_t)bh * L_ + qrowBase + mb + r) * LAT_;
        out[rowoff + n0]      = f2bf(o0[r] * inv);
        out[rowoff + 16 + n0] = f2bf(o1[r] * inv);
    }
}

// ---------- K4: latent decompress  a2[b,l,h*128+d] = attn @ Wd^T + bd ---------
__global__ void __launch_bounds__(256)
k_decompress(const __bf16* __restrict__ ob, const __bf16* __restrict__ Wd,
             const float* __restrict__ bd, __bf16* __restrict__ a2) {
    const int bh = blockIdx.y, b = bh / H_, h = bh % H_;
    const int l0 = blockIdx.x * 128;
    const int tid = threadIdx.x, lane = tid & 31, wave = tid >> 5;
    bf16x16 af = load_frag_a(&ob[((size_t)bh * L_ + l0 + wave * 16) * LAT_], LAT_);
    const int n0 = lane & 15, mb = (lane >> 4) * 8;
    #pragma unroll
    for (int j = 0; j < 8; ++j) {
        bf16x16 wf = load_frag_b(&Wd[(j * 16) * LAT_], LAT_);
        f32x8 c = {};
        c = wmma_bf16(af, wf, c);
        int gd = j * 16 + n0;
        float bv = bd[gd];
        #pragma unroll
        for (int r = 0; r < 8; ++r) {
            int l = l0 + wave * 16 + mb + r;
            a2[((size_t)(b * L_ + l)) * D_ + h * HD_ + gd] = f2bf(c[r] + bv);
        }
    }
}

// ---------- host launch -------------------------------------------------------
extern "C" void kernel_launch(void* const* d_in, const int* in_sizes, int n_in,
                              void* d_out, int out_size, void* d_ws, size_t ws_size,
                              hipStream_t stream) {
    const float* x   = (const float*)d_in[0];
    const float* fc  = (const float*)d_in[1];
    const float* Wq  = (const float*)d_in[2];   const float* bq  = (const float*)d_in[3];
    const float* Wk  = (const float*)d_in[4];   const float* bk  = (const float*)d_in[5];
    const float* Wv  = (const float*)d_in[6];   const float* bv  = (const float*)d_in[7];
    const float* Wqc = (const float*)d_in[8];   const float* bqc = (const float*)d_in[9];
    const float* Wkc = (const float*)d_in[10];  const float* bkc = (const float*)d_in[11];
    const float* Wvc = (const float*)d_in[12];  const float* bvc = (const float*)d_in[13];
    const float* Wd  = (const float*)d_in[14];  const float* bd  = (const float*)d_in[15];
    const float* Wo  = (const float*)d_in[16];  const float* bo  = (const float*)d_in[17];
    float* out = (float*)d_out;

    char* ws = (char*)d_ws;
    size_t off = 0;
    auto alloc = [&](size_t nelem) {
        __bf16* p = (__bf16*)(ws + off);
        off += ((nelem * 2 + 255) & ~(size_t)255);
        return p;
    };
    const size_t NX = (size_t)B_ * L_ * D_;        // 8M
    const size_t NW = (size_t)D_ * D_;             // 4M
    const size_t NL = (size_t)LAT_ * HD_;          // 4K
    const size_t NC = (size_t)B_ * H_ * L_ * LAT_; // 2M
    __bf16* xb   = alloc(NX);
    __bf16* Wqb  = alloc(NW); __bf16* Wkb = alloc(NW); __bf16* Wvb = alloc(NW);
    __bf16* Wob  = alloc(NW);
    __bf16* Wqcb = alloc(NL); __bf16* Wkcb = alloc(NL); __bf16* Wvcb = alloc(NL);
    __bf16* Wdb  = alloc(NL);
    __bf16* qb   = alloc(NX); __bf16* kb  = alloc(NX); __bf16* vb  = alloc(NX);
    __bf16* qcb  = alloc(NC); __bf16* kcb = alloc(NC); __bf16* vcb = alloc(NC);
    __bf16* oab  = alloc(NC);
    __bf16* a2b  = alloc(NX);

    dim3 blk(256);
    auto cvt = [&](const float* s, __bf16* d, size_t n) {
        k_convert<<<1024, blk, 0, stream>>>(s, d, (int)n);
    };
    cvt(x, xb, NX);
    cvt(Wq, Wqb, NW);   cvt(Wk, Wkb, NW);   cvt(Wv, Wvb, NW);   cvt(Wo, Wob, NW);
    cvt(Wqc, Wqcb, NL); cvt(Wkc, Wkcb, NL); cvt(Wvc, Wvcb, NL); cvt(Wd, Wdb, NL);

    const int M = B_ * L_;                          // 4096
    dim3 gGemm(M / 128, D_ / 128);                  // 32 x 16
    k_gemm<false><<<gGemm, blk, 0, stream>>>(xb, Wqb, bq, qb, M, D_, D_);
    k_gemm<false><<<gGemm, blk, 0, stream>>>(xb, Wkb, bk, kb, M, D_, D_);
    k_gemm<false><<<gGemm, blk, 0, stream>>>(xb, Wvb, bv, vb, M, D_, D_);

    dim3 gHead(L_ / 128, B_ * H_);                  // 16 x 32
    k_compress<<<gHead, blk, 0, stream>>>(qb, fc, Wqcb, bqc, qcb, 1);
    k_compress<<<gHead, blk, 0, stream>>>(kb, fc, Wkcb, bkc, kcb, 1);
    k_compress<<<gHead, blk, 0, stream>>>(vb, fc, Wvcb, bvc, vcb, 0);

    k_attn<<<gHead, blk, 0, stream>>>(qcb, kcb, vcb, oab);
    k_decompress<<<gHead, blk, 0, stream>>>(oab, Wdb, bd, a2b);

    k_gemm<true><<<gGemm, blk, 0, stream>>>(a2b, Wob, bo, (void*)out, M, D_, D_);
}
